// BaseSelfAttention_6485400617273
// MI455X (gfx1250) — matrix operand order repaired and checked by reading, not compiled, not run on hardware
//
#include <hip/hip_runtime.h>
#include <hip/hip_bf16.h>

typedef __attribute__((ext_vector_type(16))) __bf16 v16bf;
typedef __attribute__((ext_vector_type(8)))  float  v8f;

// Problem constants (from reference)
constexpr int HH  = 16;
constexpr int TT  = 4096;
constexpr int DD  = 64;
constexpr int BQ  = 128;          // query block
constexpr int GG  = 64;           // global tokens
constexpr int TSP = TT / 4;       // sparse token count = 1024
constexpr int CTX = GG + 2*128 + 3*BQ;  // 704
constexpr int CH  = 64;           // keys per chunk
constexpr int NCH = CTX / CH;     // 11
constexpr float NEGM = -1e30f;

__device__ __forceinline__ unsigned short f2bfh(float f) {
  return __builtin_bit_cast(unsigned short, (__bf16)f);   // native cvt, RNE
}

union V16U {
  unsigned short s[16];
  uint4  q[2];
  v16bf  v;
};

// 16 bf16 halfs from LDS as two 16B-aligned b128 loads
__device__ __forceinline__ v16bf ld16(const unsigned short* p0, const unsigned short* p1) {
  V16U u;
  u.q[0] = *reinterpret_cast<const uint4*>(p0);
  u.q[1] = *reinterpret_cast<const uint4*>(p1);
  return u.v;
}

// xor-shuffle within 16-lane halves; prefer VALU v_permlane16_b32 over ds_bpermute
__device__ __forceinline__ float xor16(float v, int m, int sel_lo, int sel_hi) {
#if __has_builtin(__builtin_amdgcn_permlane16)
  unsigned i = __builtin_bit_cast(unsigned, v);
  i = __builtin_amdgcn_permlane16(i, i, (unsigned)sel_lo, (unsigned)sel_hi, false, false);
  return __builtin_bit_cast(float, i);
#else
  return __shfl_xor(v, m, 16);
#endif
}
__device__ __forceinline__ float redmax16(float v) {
  v = fmaxf(v, xor16(v, 1, 0x67452301, 0xEFCDAB89));
  v = fmaxf(v, xor16(v, 2, 0x54761032, 0xDCFE98BA));
  v = fmaxf(v, xor16(v, 4, 0x32107654, 0xBA98FEDC));
  v = fmaxf(v, xor16(v, 8, 0xFEDCBA98, 0x76543210));
  return v;
}
__device__ __forceinline__ float redsum16(float v) {
  v += xor16(v, 1, 0x67452301, 0xEFCDAB89);
  v += xor16(v, 2, 0x54761032, 0xDCFE98BA);
  v += xor16(v, 4, 0x32107654, 0xBA98FEDC);
  v += xor16(v, 8, 0xFEDCBA98, 0x76543210);
  return v;
}

__global__ __launch_bounds__(256) void lsg_attn_wmma(
    const float* __restrict__ Q,   const float* __restrict__ Kl,  const float* __restrict__ Vl,
    const float* __restrict__ Ksp, const float* __restrict__ Vsp,
    const float* __restrict__ Kgl, const float* __restrict__ Vgl,
    const float* __restrict__ AM,  const float* __restrict__ SM,  const float* __restrict__ GM,
    float* __restrict__ Out)
{
  const int b   = blockIdx.x;         // 0..31 query block
  const int h   = blockIdx.y;
  const int n   = blockIdx.z;
  const int nh  = n * HH + h;
  const int tid = threadIdx.x;
  const int w    = tid >> 5;          // wave 0..7 (16 query rows each)
  const int lane = tid & 31;

  // LDS staging (double-buffered K, V^T; per-wave P tile; per-chunk mask)
  __shared__ __align__(16) unsigned short sk [2][CH][DD];   // K chunk  [key][d]   16 KB
  __shared__ __align__(16) unsigned short svT[2][DD][CH];   // V chunk  [d][key]   16 KB
  __shared__ __align__(16) unsigned short sp [8][16][CH];   // per-wave P [row][key] 16 KB
  __shared__ float smask[2][CH];

  // ---- loader roles: 4 threads per key, 16 d-elements per thread ----
  const int lkey = tid >> 2;          // 0..63
  const int ld0  = (tid & 3) * 16;    // 0,16,32,48

  auto gather = [&](int j, const float*& krow, const float*& vrow, float& mk) {
    if (j < GG) {
      size_t r = (size_t)nh * GG + j;
      krow = Kgl + r * DD; vrow = Vgl + r * DD;
      mk = GM[n * GG + j];
    } else if (j < GG + 256) {
      int sj = j - GG;
      int t  = b * 32 + (sj < 128 ? sj - 160 : sj - 64);
      if (t >= 0 && t < TSP) {
        size_t r = (size_t)nh * TSP + t;
        krow = Ksp + r * DD; vrow = Vsp + r * DD;
        mk = SM[n * TSP + t];
      } else { krow = nullptr; vrow = nullptr; mk = NEGM; }
    } else {
      int lj = j - (GG + 256);
      int t  = b * BQ - BQ + lj;
      if (t >= 0 && t < TT) {
        size_t r = (size_t)nh * TT + t;
        krow = Kl + r * DD; vrow = Vl + r * DD;
        mk = AM[n * TT + t];
      } else { krow = nullptr; vrow = nullptr; mk = NEGM; }
    }
  };

  auto load_chunk = [&](int c, int buf) {
    const float *krow, *vrow; float mk;
    gather(c * CH + lkey, krow, vrow, mk);
    float kv[16], vv[16];
    if (krow) {
      const float4* kp = reinterpret_cast<const float4*>(krow + ld0);
      const float4* vp = reinterpret_cast<const float4*>(vrow + ld0);
      #pragma unroll
      for (int g = 0; g < 4; ++g) {
        float4 k4 = kp[g], v4 = vp[g];
        kv[g*4+0]=k4.x; kv[g*4+1]=k4.y; kv[g*4+2]=k4.z; kv[g*4+3]=k4.w;
        vv[g*4+0]=v4.x; vv[g*4+1]=v4.y; vv[g*4+2]=v4.z; vv[g*4+3]=v4.w;
      }
    } else {
      #pragma unroll
      for (int i = 0; i < 16; ++i) { kv[i] = 0.f; vv[i] = 0.f; }
    }
    // K: 16 contiguous bf16 -> two b128 LDS stores
    V16U u;
    #pragma unroll
    for (int i = 0; i < 16; ++i) u.s[i] = f2bfh(kv[i]);
    *reinterpret_cast<uint4*>(&sk[buf][lkey][ld0])     = u.q[0];
    *reinterpret_cast<uint4*>(&sk[buf][lkey][ld0 + 8]) = u.q[1];
    // V transposed: scattered b16 stores
    #pragma unroll
    for (int i = 0; i < 16; ++i) svT[buf][ld0 + i][lkey] = f2bfh(vv[i]);
    if ((tid & 3) == 0) smask[buf][lkey] = mk;
  };

  // ---- per-lane WMMA layout indices ----
  const int col  = lane & 15;          // C-layout column / A-layout row
  const int kb   = (lane >> 4) * 8;    // A-operand K sub-base (0 or 8)
  const int klo  = (lane >> 4) * 16;   // unused; B operand also uses kb (0 or 8)
  const int rb   = (lane >> 4) * 8;    // C-layout row base (0 or 8)

  // ---- load Q tile (16 rows of this wave) into A-layout registers, bf16 ----
  // Pre-scale by 1/sqrt(64) = 0.125 (power of two -> exact in bf16).
  const float* qrow = Q + ((size_t)nh * TT + (size_t)b * BQ + w * 16 + col) * DD;
  v16bf qa[2];
  #pragma unroll
  for (int ch = 0; ch < 2; ++ch) {
    V16U u;
    #pragma unroll
    for (int i = 0; i < 8; ++i) u.s[i]     = f2bfh(0.125f * qrow[ch * 32 + kb + i]);
    #pragma unroll
    for (int i = 0; i < 8; ++i) u.s[8 + i] = f2bfh(0.125f * qrow[ch * 32 + kb + 16 + i]);
    qa[ch] = u.v;
  }

  // ---- flash state ----
  v8f o[4];
  float mrow[8], lrow[8];
  #pragma unroll
  for (int j = 0; j < 4; ++j) o[j] = (v8f)0.f;
  #pragma unroll
  for (int r = 0; r < 8; ++r) { mrow[r] = NEGM; lrow[r] = 0.f; }

  load_chunk(0, 0);

  for (int c = 0; c < NCH; ++c) {
    __syncthreads();
    if (c + 1 < NCH) load_chunk(c + 1, (c + 1) & 1);
    const int buf = c & 1;

    // ---- scores: S(16q x 64k) = Q(16x64) * K^T, four 16-key groups ----
    v8f s[4];
    #pragma unroll
    for (int g = 0; g < 4; ++g) s[g] = (v8f)0.f;
    #pragma unroll
    for (int dc = 0; dc < 2; ++dc) {
      #pragma unroll
      for (int g = 0; g < 4; ++g) {
        v16bf bk = ld16(&sk[buf][g * 16 + col][dc * 32 + kb],
                        &sk[buf][g * 16 + col][dc * 32 + kb + 16]);
        s[g] = __builtin_amdgcn_wmma_f32_16x16x32_bf16(false, qa[dc], false, bk,
                                                       (short)0, s[g], false, false);
      }
    }
    float mk[4];
    #pragma unroll
    for (int g = 0; g < 4; ++g) mk[g] = smask[buf][g * 16 + col];

    // ---- online softmax update (rows live across 16-lane halves) ----
    float pv[4][8];
    #pragma unroll
    for (int r = 0; r < 8; ++r) {
      float a[4];
      #pragma unroll
      for (int g = 0; g < 4; ++g) a[g] = s[g][r] + mk[g];
      float cm = fmaxf(fmaxf(a[0], a[1]), fmaxf(a[2], a[3]));
      cm = redmax16(cm);
      float mo = mrow[r];
      float mn = fmaxf(mo, cm);
      float sc = __expf(mo - mn);
      float ps = 0.f;
      #pragma unroll
      for (int g = 0; g < 4; ++g) {
        float p = __expf(a[g] - mn);
        pv[g][r] = p;
        ps += p;
      }
      ps = redsum16(ps);
      lrow[r] = lrow[r] * sc + ps;
      mrow[r] = mn;
      #pragma unroll
      for (int j = 0; j < 4; ++j) o[j][r] *= sc;
    }

    // ---- stage P (C-layout) to LDS, re-read in A-layout ----
    #pragma unroll
    for (int r = 0; r < 8; ++r) {
      #pragma unroll
      for (int g = 0; g < 4; ++g)
        sp[w][rb + r][g * 16 + col] = f2bfh(pv[g][r]);
    }
    asm volatile("s_wait_dscnt 0" ::: "memory");
    __builtin_amdgcn_wave_barrier();

    // ---- O(16x64) += P(16x64) * V(64x64): two 32-key slabs x four d slices ----
    #pragma unroll
    for (int kc = 0; kc < 2; ++kc) {
      v16bf pa = ld16(&sp[w][col][kc * 32 + kb], &sp[w][col][kc * 32 + kb + 16]);
      #pragma unroll
      for (int j = 0; j < 4; ++j) {
        v16bf bv = ld16(&svT[buf][j * 16 + col][kc * 32 + kb],
                        &svT[buf][j * 16 + col][kc * 32 + kb + 16]);
        o[j] = __builtin_amdgcn_wmma_f32_16x16x32_bf16(false, pa, false, bv,
                                                       (short)0, o[j], false, false);
      }
    }
  }

  // ---- epilogue: normalize and store ----
  float* orow = Out + ((size_t)nh * TT + (size_t)b * BQ + w * 16) * DD;
  #pragma unroll
  for (int r = 0; r < 8; ++r) {
    float inv = 1.0f / lrow[r];
    #pragma unroll
    for (int j = 0; j < 4; ++j)
      orow[(size_t)(rb + r) * DD + j * 16 + col] = o[j][r] * inv;
  }
}

extern "C" void kernel_launch(void* const* d_in, const int* in_sizes, int n_in,
                              void* d_out, int out_size, void* d_ws, size_t ws_size,
                              hipStream_t stream) {
  const float* Q   = (const float*)d_in[0];
  const float* Kl  = (const float*)d_in[1];
  const float* Vl  = (const float*)d_in[2];
  const float* Ksp = (const float*)d_in[3];
  const float* Vsp = (const float*)d_in[4];
  const float* Kgl = (const float*)d_in[5];
  const float* Vgl = (const float*)d_in[6];
  const float* AM  = (const float*)d_in[7];
  const float* SM  = (const float*)d_in[8];
  const float* GM  = (const float*)d_in[9];
  dim3 grid(32, 16, 2), blk(256);
  hipLaunchKernelGGL(lsg_attn_wmma, grid, blk, 0, stream,
                     Q, Kl, Vl, Ksp, Vsp, Kgl, Vgl, AM, SM, GM, (float*)d_out);
}
